// LLMAttention_71579924955323
// MI455X (gfx1250) — matrix operand order repaired and checked by reading, not compiled, not run on hardware
//
#include <hip/hip_runtime.h>
#include <hip/hip_bf16.h>
#include <math.h>

// ---------------------------------------------------------------------------
// HyperAttention for MI455X (gfx1250, wave32):
//  - bf16 WMMA 16x16x32 (f32 accum) for QKV GEMM, QK^T, PV, output GEMM
//  - bf16 operands pre-converted in memory (no cvt in hot loops)
//  - K/V staged to LDS via GLOBAL_LOAD_ASYNC_TO_LDS_B128 (ASYNCcnt path)
//  - stable 128-bucket counting sort in 128KB LDS (320KB WGP budget)
// ---------------------------------------------------------------------------

typedef __attribute__((ext_vector_type(16))) __bf16 v16bf;
typedef __attribute__((ext_vector_type(8)))  float  v8f;

#define B_   2
#define H_   16
#define L_   8192
#define DH_  64
#define D_   1024
#define TN_  3072
#define BLK_ 256
#define NB_  32          // L_/BLK_
#define SMP_ 256
#define SCALE_ 0.125f    // dh^-0.5 = 64^-0.5
#define LOG32_ 3.4657359027997265f   // log(L/SAMPLE) = log(32)
#define NEGBIG_ (-3.402823466e38f)   // finfo(float32).min

// ---- WMMA helpers ---------------------------------------------------------
__device__ inline v8f wmma_bf16(v16bf a, v16bf b, v8f c) {
  return __builtin_amdgcn_wmma_f32_16x16x32_bf16(false, a, false, b, (short)0, c,
                                                 false, false);
}

// A-fragment (16x32 bf16) from a bf16 row-major row pointer.
// lane<16: row=lane,     elems 0..15 <- K=kb+0..15
// lane>=16: row=lane-16, elems 0..15 <- K=kb+16..31
__device__ inline v16bf a_frag_bf16(const __bf16* __restrict__ row, int kb, int lane) {
  int kh = (lane >> 4) & 1;
  const __bf16* p0 = row + kb + kh * 16;
  const __bf16* p1 = row + kb + 8 + kh * 16;
  v16bf a;
#pragma unroll
  for (int i = 0; i < 8; i++) a[i] = p0[i];
#pragma unroll
  for (int i = 0; i < 8; i++) a[8 + i] = p1[i];
  return a;
}

// B-fragment (32x16 bf16): source stores one column's K-values contiguously
// (Bt is [N][K] row-major). lane<16: K=kb+0..15, lane>=16: K=kb+16..31.
__device__ inline v16bf b_frag_rowK(const __bf16* __restrict__ col, int kb, int lane) {
  const __bf16* p = col + kb + ((lane >> 4) & 1) * 16;
  v16bf b;
#pragma unroll
  for (int i = 0; i < 16; i++) b[i] = p[i];
  return b;
}

// Async DMA: copy 16B from global (bf16) into LDS, tracked by ASYNCcnt.
__device__ inline void async_copy_b128(const __bf16* g, void* lds) {
  unsigned ldsOff = (unsigned)(unsigned long long)lds;   // AS(3) offset in low bits
  asm volatile("global_load_async_to_lds_b128 %0, %1, off"
               :: "v"(ldsOff), "v"(g) : "memory");
}
__device__ inline void async_wait0() {
  asm volatile("s_wait_asynccnt 0x0" ::: "memory");
}

// ---- 0. Convert/transpose weights + convert x to bf16 ---------------------
__global__ __launch_bounds__(256) void prep_inputs(
    const float* __restrict__ Wqkv, const float* __restrict__ Wproj,
    const float* __restrict__ x,
    __bf16* __restrict__ Wtq, __bf16* __restrict__ Wtp, __bf16* __restrict__ xb) {
  int i = blockIdx.x * 256 + threadIdx.x;
  if (i < D_ * TN_) {
    int k = i / TN_, n = i % TN_;
    Wtq[(size_t)n * D_ + k] = (__bf16)Wqkv[i];
  }
  int j = i - D_ * TN_;
  if (j >= 0 && j < D_ * D_) {
    int k = j / D_, n = j % D_;
    Wtp[(size_t)n * D_ + k] = (__bf16)Wproj[j];
  }
  int m = j - D_ * D_;
  if (m >= 0 && m < B_ * L_ * D_) xb[m] = (__bf16)x[m];
}

// ---- 1./7. WMMA GEMM: out[M][N] = A[M][K] x Bt[N][K]^T + bias -------------
// MODE 0: plain row-major f32 store. MODE 1: scatter f32+bf16 to [t][b][h][l][64].
template <int NTOT, int KTOT, int MODE>
__global__ __launch_bounds__(256) void gemm_wmma(
    const __bf16* __restrict__ A, const __bf16* __restrict__ Bt,
    const float* __restrict__ bias, float* __restrict__ out,
    __bf16* __restrict__ outb) {
  int lane = threadIdx.x & 31;
  int wave = threadIdx.x >> 5;
  int mbase = blockIdx.x * 128 + wave * 16;
  int n0 = blockIdx.y * 64;
  const __bf16* arow = A + (size_t)(mbase + (lane & 15)) * KTOT;

  v8f acc[4] = {};
  for (int kb = 0; kb < KTOT; kb += 32) {
    v16bf a = a_frag_bf16(arow, kb, lane);
#pragma unroll
    for (int nt = 0; nt < 4; nt++) {
      const __bf16* col = Bt + (size_t)(n0 + nt * 16 + (lane & 15)) * KTOT;
      acc[nt] = wmma_bf16(a, b_frag_rowK(col, kb, lane), acc[nt]);
    }
  }
#pragma unroll
  for (int nt = 0; nt < 4; nt++) {
    int n = n0 + nt * 16 + (lane & 15);
    float bv = bias[n];
#pragma unroll
    for (int r = 0; r < 8; r++) {
      int m = mbase + r + 8 * (lane >> 4);
      float v = acc[nt][r] + bv;
      if (MODE == 0) {
        out[(size_t)m * NTOT + n] = v;
      } else {
        int bb = m / L_, ll = m % L_;
        int t = n / 1024, hh = (n / 64) & 15, dd = n & 63;
        size_t dst = ((((size_t)t * B_ + bb) * H_ + hh) * L_ + ll) * DH_ + dd;
        out[dst] = v;
        outb[dst] = (__bf16)v;
      }
    }
  }
}

// ---- 2. LSH hash (f32 for sign fidelity): bucket = gray(id) ---------------
__global__ __launch_bounds__(256) void lsh_hash(
    const float* __restrict__ qkv, const float* __restrict__ proj,
    int* __restrict__ bq, int* __restrict__ bk) {
  int i = blockIdx.x * 256 + threadIdx.x;       // over 2*B*H*L
  int which = i / (B_ * H_ * L_);
  int rem = i % (B_ * H_ * L_);
  const float* row = qkv + ((size_t)which * B_ * H_ * L_ + rem) * DH_;
  float acc[7] = {};
  for (int d = 0; d < DH_; d++) {
    float x = row[d];
#pragma unroll
    for (int r = 0; r < 7; r++) acc[r] += x * proj[d * 7 + r];
  }
  int id = 0;
#pragma unroll
  for (int r = 0; r < 7; r++) id |= (acc[r] > 0.f) ? (1 << r) : 0;
  (which ? bk : bq)[rem] = id ^ (id >> 1);      // hamming perm == Gray code
}

// ---- 3. Stable counting sort over 128 buckets, one WG per (b,h,{q|k}) -----
__global__ __launch_bounds__(256) void bucket_sort(
    const int* __restrict__ bq, const int* __restrict__ bk,
    int* __restrict__ qidx, int* __restrict__ qinv, int* __restrict__ kidx) {
  __shared__ int cnt[128 * 256];
  __shared__ int tot[128];
  __shared__ int base[128];
  int g = blockIdx.x;
  int isK = g >= (B_ * H_);
  int pair = g & (B_ * H_ - 1);
  const int* bkt = (isK ? bk : bq) + (size_t)pair * L_;
  int* sidx = (isK ? kidx : qidx) + (size_t)pair * L_;
  int* sinv = isK ? nullptr : qinv + (size_t)pair * L_;

  int t = threadIdx.x;
  for (int i = t; i < 128 * 256; i += 256) cnt[i] = 0;
  __syncthreads();
  int loc[32];
#pragma unroll
  for (int i = 0; i < 32; i++) {
    loc[i] = bkt[t * 32 + i];
    cnt[loc[i] * 256 + t]++;                    // column t owned by thread t
  }
  __syncthreads();
  if (t < 128) {                                // prefix over threads per bin
    int s = 0;
    for (int tt = 0; tt < 256; tt++) { int c = cnt[t * 256 + tt]; cnt[t * 256 + tt] = s; s += c; }
    tot[t] = s;
  }
  __syncthreads();
  if (t == 0) { int s = 0; for (int v = 0; v < 128; v++) { base[v] = s; s += tot[v]; } }
  __syncthreads();
#pragma unroll
  for (int i = 0; i < 32; i++) {
    int v = loc[i];
    int r = base[v] + cnt[v * 256 + t]++;
    int orig = t * 32 + i;
    sidx[r] = orig;
    if (sinv) sinv[orig] = r;
  }
}

// ---- shared attention core ------------------------------------------------
// Stages 256 gathered bf16 K/V rows (128B each) into LDS via async DMA.
__device__ inline void stage_kv_async(const __bf16* kb, const __bf16* vb,
                                      const int* kid, size_t bhL,
                                      __bf16 (*Ksh)[64], __bf16 (*Vsh)[64]) {
  for (int i = threadIdx.x; i < 256 * 8; i += 256) {   // (row, 16B segment)
    int row = i >> 3, seg = i & 7;
    size_t src = (bhL + kid[row]) * DH_ + seg * 8;
    async_copy_b128(kb + src, &Ksh[row][seg * 8]);
    async_copy_b128(vb + src, &Vsh[row][seg * 8]);
  }
  async_wait0();
  __syncthreads();
}

// ---- 4. Block-diagonal attention over sorted order ------------------------
__global__ __launch_bounds__(256) void block_attn(
    const __bf16* __restrict__ qkvb, const int* __restrict__ qidx,
    const int* __restrict__ kidx, float* __restrict__ ab, float* __restrict__ lseb) {
  __shared__ __align__(16) __bf16 Ksh[256][64];
  __shared__ __align__(16) __bf16 Vsh[256][64];
  __shared__ __align__(16) __bf16 Psh[8][16 * 256];
  __shared__ int kid[256];

  int qblk = blockIdx.x % NB_;
  int bh = blockIdx.x / NB_;
  size_t bhL = (size_t)bh * L_;
  const __bf16* qbase = qkvb;
  const __bf16* kbase = qkvb + (size_t)1 * B_ * H_ * L_ * DH_;
  const __bf16* vbase = qkvb + (size_t)2 * B_ * H_ * L_ * DH_;

  kid[threadIdx.x] = kidx[bhL + qblk * 256 + threadIdx.x];
  __syncthreads();
  stage_kv_async(kbase, vbase, kid, bhL, Ksh, Vsh);

  int lane = threadIdx.x & 31;
  int wave = threadIdx.x >> 5;
  __bf16* prow = &Psh[wave][0];

  for (int sub = 0; sub < 2; sub++) {
    int rowbase = sub * 128 + wave * 16;
    int qrow = qidx[bhL + qblk * 256 + rowbase + (lane & 15)];
    const __bf16* qrp = qbase + (bhL + qrow) * DH_;
    v16bf a0 = a_frag_bf16(qrp, 0, lane);
    v16bf a1 = a_frag_bf16(qrp, 32, lane);

    v8f s[16] = {};
#pragma unroll
    for (int nt = 0; nt < 16; nt++) {
      const __bf16* col = &Ksh[nt * 16 + (lane & 15)][0];
      s[nt] = wmma_bf16(a0, b_frag_rowK(col, 0, lane), s[nt]);
      s[nt] = wmma_bf16(a1, b_frag_rowK(col, 32, lane), s[nt]);
    }

    float lsum[8], lmax[8];
#pragma unroll
    for (int r = 0; r < 8; r++) {
      float m = NEGBIG_;
#pragma unroll
      for (int nt = 0; nt < 16; nt++) m = fmaxf(m, s[nt][r]);
#pragma unroll
      for (int off = 8; off >= 1; off >>= 1) m = fmaxf(m, __shfl_xor(m, off, 16));
      m *= SCALE_;
      float sum = 0.f;
      int rl = r + 8 * (lane >> 4);
#pragma unroll
      for (int nt = 0; nt < 16; nt++) {
        float e = __expf(s[nt][r] * SCALE_ - m);
        prow[rl * 256 + nt * 16 + (lane & 15)] = (__bf16)e;
        sum += e;
      }
#pragma unroll
      for (int off = 8; off >= 1; off >>= 1) sum += __shfl_xor(sum, off, 16);
      lmax[r] = m;
      lsum[r] = sum;
    }

    v8f o[4] = {};
#pragma unroll
    for (int kc = 0; kc < 8; kc++) {
      int kh = (lane >> 4) & 1;
      const __bf16* p0 = prow + (lane & 15) * 256 + kc * 32 + kh * 16;
      v16bf pa;
#pragma unroll
      for (int i = 0; i < 8; i++) pa[i] = p0[i];
#pragma unroll
      for (int i = 0; i < 8; i++) pa[8 + i] = p0[8 + i];
#pragma unroll
      for (int dt = 0; dt < 4; dt++) {
        const __bf16* vp = &Vsh[kc * 32 + kh * 16][dt * 16 + (lane & 15)];
        v16bf vb;
#pragma unroll
        for (int i = 0; i < 16; i++) vb[i] = vp[i * 64];
        o[dt] = wmma_bf16(pa, vb, o[dt]);
      }
    }

#pragma unroll
    for (int r = 0; r < 8; r++) {
      int p = qblk * 256 + rowbase + r + 8 * (lane >> 4);
      float rs = 1.0f / lsum[r];
#pragma unroll
      for (int dt = 0; dt < 4; dt++)
        ab[(bhL + p) * DH_ + dt * 16 + (lane & 15)] = o[dt][r] * rs;
      if ((lane & 15) == 0) lseb[bhL + p] = lmax[r] + __logf(lsum[r]);
    }
  }
}

// ---- 5. Residual sampled attention with block-collision mask --------------
__global__ __launch_bounds__(256) void samp_attn(
    const __bf16* __restrict__ qkvb, const int* __restrict__ qidx,
    const int* __restrict__ kidx, const int* __restrict__ sset,
    float* __restrict__ ar, float* __restrict__ lser) {
  __shared__ __align__(16) __bf16 Ksh[256][64];
  __shared__ __align__(16) __bf16 Vsh[256][64];
  __shared__ __align__(16) __bf16 Psh[8][16 * 256];
  __shared__ int bidArr[256];
  __shared__ int kid[256];

  int qblk = blockIdx.x % NB_;     // query block id (p >> 8)
  int bh = blockIdx.x / NB_;
  size_t bhL = (size_t)bh * L_;
  const __bf16* qbase = qkvb;
  const __bf16* kbase = qkvb + (size_t)1 * B_ * H_ * L_ * DH_;
  const __bf16* vbase = qkvb + (size_t)2 * B_ * H_ * L_ * DH_;

  {
    int sp = sset[bh * SMP_ + threadIdx.x];
    bidArr[threadIdx.x] = sp >> 8;
    kid[threadIdx.x] = kidx[bhL + sp];
  }
  __syncthreads();
  stage_kv_async(kbase, vbase, kid, bhL, Ksh, Vsh);

  int lane = threadIdx.x & 31;
  int wave = threadIdx.x >> 5;
  __bf16* prow = &Psh[wave][0];

  float bias[16];
#pragma unroll
  for (int nt = 0; nt < 16; nt++)
    bias[nt] = (bidArr[nt * 16 + (lane & 15)] == qblk) ? NEGBIG_ : 0.f;

  for (int sub = 0; sub < 2; sub++) {
    int rowbase = sub * 128 + wave * 16;
    int qrow = qidx[bhL + qblk * 256 + rowbase + (lane & 15)];
    const __bf16* qrp = qbase + (bhL + qrow) * DH_;
    v16bf a0 = a_frag_bf16(qrp, 0, lane);
    v16bf a1 = a_frag_bf16(qrp, 32, lane);

    v8f s[16] = {};
#pragma unroll
    for (int nt = 0; nt < 16; nt++) {
      const __bf16* col = &Ksh[nt * 16 + (lane & 15)][0];
      s[nt] = wmma_bf16(a0, b_frag_rowK(col, 0, lane), s[nt]);
      s[nt] = wmma_bf16(a1, b_frag_rowK(col, 32, lane), s[nt]);
    }

    float lsum[8], lmax[8];
#pragma unroll
    for (int r = 0; r < 8; r++) {
      float m = NEGBIG_;
#pragma unroll
      for (int nt = 0; nt < 16; nt++) m = fmaxf(m, s[nt][r] * SCALE_ + bias[nt]);
#pragma unroll
      for (int off = 8; off >= 1; off >>= 1) m = fmaxf(m, __shfl_xor(m, off, 16));
      float sum = 0.f;
      int rl = r + 8 * (lane >> 4);
#pragma unroll
      for (int nt = 0; nt < 16; nt++) {
        float e = __expf(s[nt][r] * SCALE_ + bias[nt] - m);
        prow[rl * 256 + nt * 16 + (lane & 15)] = (__bf16)e;
        sum += e;
      }
#pragma unroll
      for (int off = 8; off >= 1; off >>= 1) sum += __shfl_xor(sum, off, 16);
      lmax[r] = m;
      lsum[r] = sum;
    }

    v8f o[4] = {};
#pragma unroll
    for (int kc = 0; kc < 8; kc++) {
      int kh = (lane >> 4) & 1;
      const __bf16* p0 = prow + (lane & 15) * 256 + kc * 32 + kh * 16;
      v16bf pa;
#pragma unroll
      for (int i = 0; i < 8; i++) pa[i] = p0[i];
#pragma unroll
      for (int i = 0; i < 8; i++) pa[8 + i] = p0[8 + i];
#pragma unroll
      for (int dt = 0; dt < 4; dt++) {
        const __bf16* vp = &Vsh[kc * 32 + kh * 16][dt * 16 + (lane & 15)];
        v16bf vb;
#pragma unroll
        for (int i = 0; i < 16; i++) vb[i] = vp[i * 64];
        o[dt] = wmma_bf16(pa, vb, o[dt]);
      }
    }

#pragma unroll
    for (int r = 0; r < 8; r++) {
      int p = qblk * 256 + rowbase + r + 8 * (lane >> 4);
      float rs = 1.0f / lsum[r];
#pragma unroll
      for (int dt = 0; dt < 4; dt++)
        ar[(bhL + p) * DH_ + dt * 16 + (lane & 15)] = o[dt][r] * rs;
      if ((lane & 15) == 0) lser[bhL + p] = lmax[r] + __logf(lsum[r]) + LOG32_;
    }
  }
}

// ---- 6. LSE combine + un-sort scatter to bf16 [B,L,H*dh] ------------------
__global__ __launch_bounds__(256) void combine_unsort(
    const float* __restrict__ ab, const float* __restrict__ lseb,
    const float* __restrict__ ar, const float* __restrict__ lser,
    const int* __restrict__ qidx, __bf16* __restrict__ yb) {
  int i = blockIdx.x * 256 + threadIdx.x;   // over B*H*L
  int bh = i / L_;
  int b = bh / H_, h = bh % H_;
  float c = 1.f / (1.f + __expf(lser[i] - lseb[i]));
  float cc = 1.f - c;
  int orig = qidx[i];
  const float* pa = ab + (size_t)i * DH_;
  const float* pr = ar + (size_t)i * DH_;
  __bf16* py = yb + ((size_t)b * L_ + orig) * D_ + h * DH_;
#pragma unroll
  for (int d = 0; d < DH_; d++) py[d] = (__bf16)(c * pa[d] + cc * pr[d]);
}

// ---------------------------------------------------------------------------
extern "C" void kernel_launch(void* const* d_in, const int* in_sizes, int n_in,
                              void* d_out, int out_size, void* d_ws, size_t ws_size,
                              hipStream_t stream) {
  const float* x       = (const float*)d_in[0];
  const float* W_qkv   = (const float*)d_in[1];
  const float* b_qkv   = (const float*)d_in[2];
  const float* W_proj  = (const float*)d_in[3];
  const float* b_proj  = (const float*)d_in[4];
  const float* projdir = (const float*)d_in[5];
  const int*   sset    = (const int*)d_in[6];
  float* out = (float*)d_out;

  char* w = (char*)d_ws;
  size_t off = 0;
  auto alloc = [&](size_t bytes) -> void* {
    void* p = w + off;
    off = (off + bytes + 255) & ~(size_t)255;
    return p;
  };
  const size_t BHL = (size_t)B_ * H_ * L_;
  float*  qkv  = (float*)alloc(3 * BHL * DH_ * 4);   // f32 (LSH hash source)
  __bf16* qkvb = (__bf16*)alloc(3 * BHL * DH_ * 2);  // bf16 (attention operands)
  int* bq      = (int*)alloc(BHL * 4);
  int* bk      = (int*)alloc(BHL * 4);
  int* qidx    = (int*)alloc(BHL * 4);
  int* kidx    = (int*)alloc(BHL * 4);
  int* qinv    = (int*)alloc(BHL * 4);
  float* ab    = (float*)alloc(BHL * DH_ * 4);
  float* lseb  = (float*)alloc(BHL * 4);
  float* ar    = (float*)alloc(BHL * DH_ * 4);
  float* lser  = (float*)alloc(BHL * 4);
  __bf16* yb   = (__bf16*)alloc((size_t)B_ * L_ * D_ * 2);
  __bf16* xb   = (__bf16*)alloc((size_t)B_ * L_ * D_ * 2);
  __bf16* Wtq  = (__bf16*)alloc((size_t)D_ * TN_ * 2);
  __bf16* Wtp  = (__bf16*)alloc((size_t)D_ * D_ * 2);
  (void)qinv; (void)in_sizes; (void)n_in; (void)out_size; (void)ws_size;

  // 0. one-time bf16 conversion of x and transposed weights
  int prepN = D_ * TN_ + D_ * D_ + B_ * L_ * D_;
  prep_inputs<<<(prepN + 255) / 256, 256, 0, stream>>>(W_qkv, W_proj, x, Wtq, Wtp, xb);
  // 1. QKV projection (WMMA), scatter f32 + bf16 copies to [t][b][h][l][64]
  gemm_wmma<TN_, D_, 1><<<dim3((B_ * L_) / 128, TN_ / 64), 256, 0, stream>>>(
      xb, Wtq, b_qkv, qkv, qkvb);
  // 2. LSH bucket ids for q and k (f32 for sign fidelity)
  lsh_hash<<<(2 * BHL) / 256, 256, 0, stream>>>(qkv, projdir, bq, bk);
  // 3. stable counting sort per (b,h) for q and k
  bucket_sort<<<2 * B_ * H_, 256, 0, stream>>>(bq, bk, qidx, qinv, kidx);
  // 4. block-diagonal attention (async K/V staging + WMMA)
  block_attn<<<B_ * H_ * NB_, 256, 0, stream>>>(qkvb, qidx, kidx, ab, lseb);
  // 5. sampled residual attention
  samp_attn<<<B_ * H_ * NB_, 256, 0, stream>>>(qkvb, qidx, kidx, sset, ar, lser);
  // 6. combine + un-sort (emit bf16 GEMM operand)
  combine_unsort<<<BHL / 256, 256, 0, stream>>>(ab, lseb, ar, lser, qidx, yb);
  // 7. output projection (WMMA)
  gemm_wmma<D_, D_, 0><<<dim3((B_ * L_) / 128, D_ / 64), 256, 0, stream>>>(
      yb, Wtp, b_proj, out, nullptr);
}
